// SpikingGenerator_4140348473521
// MI455X (gfx1250) — matrix operand order repaired and checked by reading, not compiled, not run on hardware
//
#include <hip/hip_runtime.h>

typedef __attribute__((ext_vector_type(16))) __bf16        v16bf;
typedef __attribute__((ext_vector_type(8)))  float         v8f;
typedef __attribute__((ext_vector_type(4)))  unsigned int  v4u;

#define B_ 256
#define N_ 1024
#define H_ 4096
#define T_ 128

// ---------------------------------------------------------------------------
// WMMA operand loaders (bf16, 16x16x32), per CDNA5 ISA 7.12.2 layouts.
// A (16x32 MxK): lanes 0-15 -> M=0..15, K chunks {0..7} and {16..23};
//                lanes 16-31 -> same M, K chunks {8..15} and {24..31}.
// B (32x16 KxN): lane&15 -> N column, K chunks same as A ({0..7},{16..23} / {8..15},{24..31}),
//                (sourced from row-major W[n][k] since B = Wtile^T).
// ---------------------------------------------------------------------------
static __device__ __forceinline__ v16bf load_a(const unsigned short* __restrict__ base,
                                               int ld, int row0, int k0, int lane) {
  const int r  = row0 + (lane & 15);
  const int kh = (lane >> 4) << 3;          // 0 or 8
  const unsigned short* p = base + (size_t)r * ld + (k0 + kh);
  union { v16bf v; v4u q[2]; } u;
  u.q[0] = *(const v4u*)(p);
  u.q[1] = *(const v4u*)(p + 16);
  return u.v;
}

static __device__ __forceinline__ v16bf load_b(const unsigned short* __restrict__ base,
                                               int ld, int row0, int k0, int lane) {
  const int r  = row0 + (lane & 15);
  const int kh = (lane >> 4) << 3;          // 0 or 8
  const unsigned short* p = base + (size_t)r * ld + (k0 + kh);
  union { v16bf v; v4u q[2]; } u;
  u.q[0] = *(const v4u*)(p);
  u.q[1] = *(const v4u*)(p + 16);
  return u.v;
}

static __device__ __forceinline__ v8f wmma_bf16(v16bf a, v16bf b, v8f c) {
  // 8 args: (neg_a, A, neg_b, B, c_mod, C, reuse_a, reuse_b)
  return __builtin_amdgcn_wmma_f32_16x16x32_bf16(false, a, false, b, (short)0, c,
                                                 false, false);
}

// ---------------------------------------------------------------------------
// Prep kernels
// ---------------------------------------------------------------------------
__global__ void cvt_f32_bf16(const float* __restrict__ src,
                             unsigned short* __restrict__ dst, int n) {
  int i = blockIdx.x * blockDim.x + threadIdx.x;
  int stride = gridDim.x * blockDim.x;
  for (; i < n; i += stride) {
    unsigned int u = __float_as_uint(src[i]);
    dst[i] = (unsigned short)((u + 0x7FFFu + ((u >> 16) & 1u)) >> 16);  // RNE
  }
}

__global__ void add_bias(const float* __restrict__ a, const float* __restrict__ b,
                         float* __restrict__ c, int n) {
  int i = blockIdx.x * blockDim.x + threadIdx.x;
  if (i < n) c[i] = a[i] + b[i];
}

__global__ void fill_u32(unsigned int* __restrict__ p, unsigned int v, int n) {
  int i = blockIdx.x * blockDim.x + threadIdx.x;
  int stride = gridDim.x * blockDim.x;
  for (; i < n; i += stride) p[i] = v;
}

// ---------------------------------------------------------------------------
// Layer 1: cur1 = inp@W1^T + spk1_in@Wr^T + (b1+br); LIF -> mem1, spk1_out.
// Block: 256 threads = 8 waves; wave tile 16(M) x 64(H); block tile 32 x 256.
// Grid: (H/256, B/32) = (16, 8)
// ---------------------------------------------------------------------------
__global__ __launch_bounds__(256) void snn_layer1(
    const unsigned short* __restrict__ inp,      // [B,N] bf16
    const unsigned short* __restrict__ spk1_in,  // [B,H] bf16
    const unsigned short* __restrict__ W1b,      // [H,N] bf16
    const unsigned short* __restrict__ Wrb,      // [H,H] bf16
    const float* __restrict__ bias1,             // [H] = b1+br
    float* __restrict__ mem1,                    // [B,H]
    unsigned short* __restrict__ spk1_out)       // [B,H] bf16
{
  const int lane = threadIdx.x & 31;
  const int wave = threadIdx.x >> 5;
  const int m0 = blockIdx.y * 32 + (wave & 1) * 16;
  const int h0 = blockIdx.x * 256 + (wave >> 1) * 64;

  v8f acc[4] = {};

  for (int k0 = 0; k0 < N_; k0 += 32) {
    v16bf a = load_a(inp, N_, m0, k0, lane);
#pragma unroll
    for (int t = 0; t < 4; ++t) {
      v16bf b = load_b(W1b, N_, h0 + t * 16, k0, lane);
      acc[t] = wmma_bf16(a, b, acc[t]);
    }
  }
  for (int k0 = 0; k0 < H_; k0 += 32) {
    v16bf a = load_a(spk1_in, H_, m0, k0, lane);
#pragma unroll
    for (int t = 0; t < 4; ++t) {
      v16bf b = load_b(Wrb, H_, h0 + t * 16, k0, lane);
      acc[t] = wmma_bf16(a, b, acc[t]);
    }
  }

  // Epilogue: C/D layout -> lane&15 = column (h), VGPR j (+8 for hi half) = row (b)
  const int mlo = (lane >> 4) << 3;
#pragma unroll
  for (int t = 0; t < 4; ++t) {
    const int h = h0 + t * 16 + (lane & 15);
    const float bsum = bias1[h];
#pragma unroll
    for (int j = 0; j < 8; ++j) {
      const int brow = m0 + mlo + j;
      const size_t idx = (size_t)brow * H_ + h;
      const float mo = mem1[idx];
      const float cur = acc[t][j] + bsum;
      const float reset = (mo > 1.0f) ? 1.0f : 0.0f;   // THRESH=1, detached
      const float mn = mo + cur - reset;               // beta = 1
      mem1[idx] = mn;
      spk1_out[idx] = (mn > 1.0f) ? (unsigned short)0x3F80u : (unsigned short)0u;
    }
  }
}

// ---------------------------------------------------------------------------
// Layer 2: cur2 = spk1@W2^T + b2; LIF -> mem2, spk2 (to output + next input).
// Block: 256 threads = 8 waves; wave tile 16(M) x 32(N); block tile 32 x 128.
// Grid: (N/128, B/32) = (8, 8)
// ---------------------------------------------------------------------------
__global__ __launch_bounds__(256) void snn_layer2(
    const unsigned short* __restrict__ spk1,     // [B,H] bf16
    const unsigned short* __restrict__ W2b,      // [N,H] bf16
    const float* __restrict__ b2,                // [N]
    float* __restrict__ mem2,                    // [B,N]
    float* __restrict__ out,                     // [B,T,N] fp32
    unsigned short* __restrict__ inp_next,       // [B,N] bf16
    int step)
{
  const int lane = threadIdx.x & 31;
  const int wave = threadIdx.x >> 5;
  const int m0 = blockIdx.y * 32 + (wave & 1) * 16;
  const int n0 = blockIdx.x * 128 + (wave >> 1) * 32;

  v8f acc[2] = {};

  for (int k0 = 0; k0 < H_; k0 += 32) {
    v16bf a = load_a(spk1, H_, m0, k0, lane);
#pragma unroll
    for (int t = 0; t < 2; ++t) {
      v16bf b = load_b(W2b, H_, n0 + t * 16, k0, lane);
      acc[t] = wmma_bf16(a, b, acc[t]);
    }
  }

  const int mlo = (lane >> 4) << 3;
#pragma unroll
  for (int t = 0; t < 2; ++t) {
    const int n = n0 + t * 16 + (lane & 15);
    const float bb = b2[n];
#pragma unroll
    for (int j = 0; j < 8; ++j) {
      const int brow = m0 + mlo + j;
      const size_t idx = (size_t)brow * N_ + n;
      const float mo = mem2[idx];
      const float cur = acc[t][j] + bb;
      const float reset = (mo > 1.0f) ? 1.0f : 0.0f;
      const float mn = mo + cur - reset;
      mem2[idx] = mn;
      const float s = (mn > 1.0f) ? 1.0f : 0.0f;
      out[(size_t)brow * ((size_t)T_ * N_) + (size_t)step * N_ + n] = s;
      inp_next[idx] = (mn > 1.0f) ? (unsigned short)0x3F80u : (unsigned short)0u;
    }
  }
}

// ---------------------------------------------------------------------------
// Host launcher
// ---------------------------------------------------------------------------
extern "C" void kernel_launch(void* const* d_in, const int* in_sizes, int n_in,
                              void* d_out, int out_size, void* d_ws, size_t ws_size,
                              hipStream_t stream) {
  const float* x  = (const float*)d_in[0];  // [B,N]
  const float* W1 = (const float*)d_in[1];  // [H,N]
  const float* b1 = (const float*)d_in[2];  // [H]
  const float* Wr = (const float*)d_in[3];  // [H,H]
  const float* br = (const float*)d_in[4];  // [H]
  const float* W2 = (const float*)d_in[5];  // [N,H]
  const float* b2 = (const float*)d_in[6];  // [N]
  // d_in[7] = num_steps (=128, compile-time T_)

  // Workspace carve-out (bf16 weights resident in L2 across all 128 steps)
  char* ws = (char*)d_ws;
  size_t off = 0;
  auto carve = [&](size_t bytes) -> char* {
    char* p = ws + off;
    off += (bytes + 255) & ~(size_t)255;
    return p;
  };
  unsigned short* W1b   = (unsigned short*)carve((size_t)H_ * N_ * 2);  // 8 MiB
  unsigned short* Wrb   = (unsigned short*)carve((size_t)H_ * H_ * 2);  // 32 MiB
  unsigned short* W2b   = (unsigned short*)carve((size_t)N_ * H_ * 2);  // 8 MiB
  float*          bias1 = (float*)         carve((size_t)H_ * 4);
  unsigned short* inp   = (unsigned short*)carve((size_t)B_ * N_ * 2);
  unsigned short* s1a   = (unsigned short*)carve((size_t)B_ * H_ * 2);
  unsigned short* s1b   = (unsigned short*)carve((size_t)B_ * H_ * 2);
  float*          mem1  = (float*)         carve((size_t)B_ * H_ * 4);
  float*          mem2  = (float*)         carve((size_t)B_ * N_ * 4);

  const int TPB = 256;
  auto gsz = [](int n) { int b = (n + 256 * 8 - 1) / (256 * 8); return b > 0 ? b : 1; };

  // --- prep (every call: deterministic under graph replay) ---
  cvt_f32_bf16<<<gsz(H_ * N_), TPB, 0, stream>>>(W1, W1b, H_ * N_);
  cvt_f32_bf16<<<gsz(H_ * H_), TPB, 0, stream>>>(Wr, Wrb, H_ * H_);
  cvt_f32_bf16<<<gsz(N_ * H_), TPB, 0, stream>>>(W2, W2b, N_ * H_);
  cvt_f32_bf16<<<gsz(B_ * N_), TPB, 0, stream>>>(x, inp, B_ * N_);
  add_bias<<<(H_ + TPB - 1) / TPB, TPB, 0, stream>>>(b1, br, bias1, H_);
  fill_u32<<<gsz(B_ * H_), TPB, 0, stream>>>((unsigned int*)mem1, 0u, B_ * H_);
  fill_u32<<<gsz(B_ * N_), TPB, 0, stream>>>((unsigned int*)mem2, 0u, B_ * N_);
  fill_u32<<<gsz(B_ * H_ / 2), TPB, 0, stream>>>((unsigned int*)s1a, 0u, B_ * H_ / 2);

  // --- time loop: 2 kernels per step, spk1 ping-pong to avoid intra-grid races ---
  dim3 g1(H_ / 256, B_ / 32);   // (16, 8)
  dim3 g2(N_ / 128, B_ / 32);   // (8, 8)
  float* out = (float*)d_out;

  for (int t = 0; t < T_; ++t) {
    unsigned short* s1_rd = (t & 1) ? s1b : s1a;
    unsigned short* s1_wr = (t & 1) ? s1a : s1b;
    snn_layer1<<<g1, TPB, 0, stream>>>(inp, s1_rd, W1b, Wrb, bias1, mem1, s1_wr);
    snn_layer2<<<g2, TPB, 0, stream>>>(s1_wr, W2b, b2, mem2, out, inp, t);
  }
}
